// CharLSTM_83614423319297
// MI455X (gfx1250) — compile-verified, matrix operand order checked by reading, not run on hardware
//
#include <hip/hip_runtime.h>

#define SEQL   1024
#define BATCH  64
#define HID    512
#define EMBD   256
#define VOCAB  512
#define NBLK   32          // recurrent grid: 32 blocks (one hidden 16-slice each)

typedef __attribute__((ext_vector_type(16))) __bf16 v16bf;
typedef __attribute__((ext_vector_type(8)))  float  v8f;

union BV { v16bf v; uint4 q[2]; };

__device__ __forceinline__ unsigned short f2bf(float f) {
    unsigned u = __builtin_bit_cast(unsigned, f);
    u = (u + 0x7FFFu + ((u >> 16) & 1u)) >> 16;
    return (unsigned short)u;
}

__device__ __forceinline__ float sigm(float x) {
    return 1.0f / (1.0f + __expf(-x));
}
__device__ __forceinline__ float tanh_f(float x) {
    float ax = fabsf(x);
    float e  = __expf(-2.0f * ax);
    float t  = (1.0f - e) / (1.0f + e);
    return copysignf(t, x);
}

__device__ __forceinline__ v8f wmma_bf16(v16bf a, v16bf b, v8f c) {
    return __builtin_amdgcn_wmma_f32_16x16x32_bf16(false, a, false, b, (short)0, c, false, false);
}

// A-operand (and row-major B=W^T operand) load in the documented 16-bit lane layout:
// lane L holds row (rowbase + L%16); lanes<16 get K = kb*32 + {0..7,16..23},
// lanes>=16 get K = kb*32 + {8..15,24..31} -> two contiguous 16B chunks per lane.
__device__ __forceinline__ v16bf load_mat(const unsigned short* __restrict__ base,
                                          int stride, int rowbase, int lane, int kb) {
    int row = rowbase + (lane & 15);
    int k0  = kb * 32 + ((lane & 16) ? 8 : 0);
    const unsigned short* p = base + (size_t)row * stride + k0;
    BV u;
    u.q[0] = *(const uint4*)(p);
    u.q[1] = *(const uint4*)(p + 16);
    return u.v;
}

__device__ __forceinline__ v16bf load_b_lds(const unsigned short* sm, int lane) {
    BV u;
    const uint4* p = (const uint4*)(sm + lane * 16);
    u.q[0] = p[0];
    u.q[1] = p[1];
    return u.v;
}

// ---------------------------------------------------------------------------
// Kernel 1: prep — barrier init, bias fuse, Wfc->bf16, state seed, emb gather
// ---------------------------------------------------------------------------
extern "C" __global__ void lstm_prep(
    const int* __restrict__ x, const float* __restrict__ h0,
    const float* __restrict__ emb,
    const float* __restrict__ bih1, const float* __restrict__ bhh1,
    const float* __restrict__ bih2, const float* __restrict__ bhh2,
    const float* __restrict__ Wfc,
    unsigned short* __restrict__ xe, unsigned short* __restrict__ h1s,
    unsigned short* __restrict__ h2s, unsigned short* __restrict__ wfcb,
    float* __restrict__ b1, float* __restrict__ b2, int* __restrict__ bar)
{
    size_t tid = (size_t)blockIdx.x * blockDim.x + threadIdx.x;
    size_t nth = (size_t)gridDim.x * blockDim.x;
    if (tid == 0) { bar[0] = 0; bar[1] = 0; }
    for (size_t i = tid; i < 4 * HID; i += nth) {
        b1[i] = bih1[i] + bhh1[i];
        b2[i] = bih2[i] + bhh2[i];
    }
    for (size_t i = tid; i < (size_t)VOCAB * HID; i += nth)
        wfcb[i] = f2bf(Wfc[i]);
    for (size_t i = tid; i < (size_t)BATCH * HID; i += nth) {
        unsigned short v = f2bf(h0[i]);
        h1s[i] = v; h1s[BATCH * HID + i] = v;
        h2s[i] = v; h2s[BATCH * HID + i] = v;
    }
    for (size_t i = tid; i < (size_t)SEQL * BATCH * EMBD; i += nth) {
        size_t row = i >> 8;          // /EMBD
        int    e   = (int)(i & (EMBD - 1));
        int    tok = x[row];
        xe[i] = f2bf(emb[(size_t)tok * EMBD + e]);
    }
}

// ---------------------------------------------------------------------------
// Kernel 2: persistent 2-layer LSTM recurrence. 32 blocks x 128 threads.
// Dynamic LDS 224KB holds this block's bf16 weight tiles, pre-swizzled.
// Pipelined: slot t computes L1(t) and L2(t-1); one grid barrier per slot.
// ---------------------------------------------------------------------------
extern "C" __global__ __launch_bounds__(128, 1) void lstm_recurrent(
    const float* __restrict__ c0, const unsigned short* __restrict__ xe,
    const float* __restrict__ Wih1, const float* __restrict__ Whh1,
    const float* __restrict__ Wih2, const float* __restrict__ Whh2,
    const float* __restrict__ b1, const float* __restrict__ b2,
    unsigned short* __restrict__ h1s, unsigned short* __restrict__ h2s,
    unsigned short* __restrict__ h2seq, int* __restrict__ bar)
{
    extern __shared__ unsigned short smem[];
    // LDS regions (bf16 elems): w1x [4][8][32][16]  @0      (K=256, Wih1)
    //                           w1h [4][16][32][16] @16384  (K=512, Whh1)
    //                           w2  [4][32][32][16] @49152  (K=1024, [Wih2|Whh2])
    const int LW1X = 0, LW1H = 16384, LW2 = 49152;
    const int tid  = threadIdx.x;
    const int lane = tid & 31;
    const int wave = tid >> 5;           // m-tile 0..3 (batch rows)
    const int jt   = blockIdx.x;         // hidden slice 0..31

    // ---- one-time: convert + swizzle weight tiles into LDS ----
    for (int idx = tid; idx < 114688; idx += 128) {
        int r = idx, g, kb, ln, e;
        if (r < 16384) {
            g = r >> 12; r &= 4095; kb = r >> 9; r &= 511; ln = r >> 4; e = r & 15;
            int row = g * HID + jt * 16 + (ln & 15);
            int k   = kb * 32 + ((ln < 16) ? 0 : 8) + ((e < 8) ? e : (8 + e));
            smem[idx] = f2bf(Wih1[(size_t)row * EMBD + k]);
        } else if (r < 49152) {
            r -= 16384;
            g = r >> 13; r &= 8191; kb = r >> 9; r &= 511; ln = r >> 4; e = r & 15;
            int row = g * HID + jt * 16 + (ln & 15);
            int k   = kb * 32 + ((ln < 16) ? 0 : 8) + ((e < 8) ? e : (8 + e));
            smem[idx] = f2bf(Whh1[(size_t)row * HID + k]);
        } else {
            r -= 49152;
            g = r >> 14; r &= 16383; kb = r >> 9; r &= 511; ln = r >> 4; e = r & 15;
            int row = g * HID + jt * 16 + (ln & 15);
            int k   = kb * 32 + ((ln < 16) ? 0 : 8) + ((e < 8) ? e : (8 + e));
            smem[idx] = (k < HID) ? f2bf(Wih2[(size_t)row * HID + k])
                                  : f2bf(Whh2[(size_t)row * HID + (k - HID)]);
        }
    }
    __syncthreads();

    // register <-> matrix mapping for C/D tiles (VGPR r, lane L):
    // row m = 16*wave + r + 8*(L/16), col n = 16*jt + L%16
    const int mrow0 = wave * 16 + ((lane >> 4) << 3);
    const int ncol  = jt * 16 + (lane & 15);

    float bi1[4], bi2[4];
#pragma unroll
    for (int g = 0; g < 4; ++g) {
        bi1[g] = b1[g * HID + ncol];
        bi2[g] = b2[g * HID + ncol];
    }
    v8f c1, c2;
#pragma unroll
    for (int r = 0; r < 8; ++r) {
        float cv = c0[(size_t)(mrow0 + r) * HID + ncol];
        c1[r] = cv; c2[r] = cv;
    }

    int epoch = 0;
    for (int t = 0; t <= SEQL; ++t) {
        // ---------------- layer 1, step t ----------------
        if (t < SEQL) {
            v8f acc[4];
#pragma unroll
            for (int g = 0; g < 4; ++g)
#pragma unroll
                for (int r = 0; r < 8; ++r) acc[g][r] = bi1[g];

            const unsigned short* xrow = xe + (size_t)t * BATCH * EMBD;
#pragma unroll 2
            for (int kb = 0; kb < 8; ++kb) {                 // x @ Wih1^T, K=256
                v16bf a = load_mat(xrow, EMBD, wave * 16, lane, kb);
#pragma unroll
                for (int g = 0; g < 4; ++g)
                    acc[g] = wmma_bf16(a, load_b_lds(smem + LW1X + (g * 8 + kb) * 512, lane), acc[g]);
            }
            const unsigned short* hp = h1s + (size_t)((t + 1) & 1) * BATCH * HID;
#pragma unroll 2
            for (int kb = 0; kb < 16; ++kb) {                // h1 @ Whh1^T, K=512
                v16bf a = load_mat(hp, HID, wave * 16, lane, kb);
#pragma unroll
                for (int g = 0; g < 4; ++g)
                    acc[g] = wmma_bf16(a, load_b_lds(smem + LW1H + (g * 16 + kb) * 512, lane), acc[g]);
            }
            unsigned short* hout = h1s + (size_t)(t & 1) * BATCH * HID;
#pragma unroll
            for (int r = 0; r < 8; ++r) {
                float ig = sigm(acc[0][r]), fg = sigm(acc[1][r]);
                float gg = tanh_f(acc[2][r]), og = sigm(acc[3][r]);
                float cn = fg * c1[r] + ig * gg;
                c1[r] = cn;
                hout[(size_t)(mrow0 + r) * HID + ncol] = f2bf(og * tanh_f(cn));
            }
        }
        // ---------------- layer 2, step t-1 ----------------
        if (t >= 1) {
            const int s = t - 1;
            v8f acc[4];
#pragma unroll
            for (int g = 0; g < 4; ++g)
#pragma unroll
                for (int r = 0; r < 8; ++r) acc[g][r] = bi2[g];

            const unsigned short* h1in = h1s + (size_t)(s & 1) * BATCH * HID;
#pragma unroll 2
            for (int kb = 0; kb < 16; ++kb) {                // h1 @ Wih2^T
                v16bf a = load_mat(h1in, HID, wave * 16, lane, kb);
#pragma unroll
                for (int g = 0; g < 4; ++g)
                    acc[g] = wmma_bf16(a, load_b_lds(smem + LW2 + (g * 32 + kb) * 512, lane), acc[g]);
            }
            const unsigned short* h2in = h2s + (size_t)((s + 1) & 1) * BATCH * HID;
#pragma unroll 2
            for (int kb = 0; kb < 16; ++kb) {                // h2 @ Whh2^T
                v16bf a = load_mat(h2in, HID, wave * 16, lane, kb);
#pragma unroll
                for (int g = 0; g < 4; ++g)
                    acc[g] = wmma_bf16(a, load_b_lds(smem + LW2 + (g * 32 + 16 + kb) * 512, lane), acc[g]);
            }
            unsigned short* hout = h2s + (size_t)(s & 1) * BATCH * HID;
            unsigned short* hsq  = h2seq + (size_t)s * BATCH * HID;
#pragma unroll
            for (int r = 0; r < 8; ++r) {
                float ig = sigm(acc[0][r]), fg = sigm(acc[1][r]);
                float gg = tanh_f(acc[2][r]), og = sigm(acc[3][r]);
                float cn = fg * c2[r] + ig * gg;
                c2[r] = cn;
                unsigned short hb = f2bf(og * tanh_f(cn));
                size_t off = (size_t)(mrow0 + r) * HID + ncol;
                hout[off] = hb;
                hsq[off]  = hb;
            }
        }
        // ---------------- grid barrier (epoch, agent scope) ----------------
        ++epoch;
        __threadfence();
        __syncthreads();
        if (tid == 0) {
            if (atomicAdd(&bar[0], 1) == NBLK - 1) {
                atomicExch(&bar[0], 0);
                __hip_atomic_store(&bar[1], epoch, __ATOMIC_RELEASE, __HIP_MEMORY_SCOPE_AGENT);
            } else {
                while (__hip_atomic_load(&bar[1], __ATOMIC_ACQUIRE, __HIP_MEMORY_SCOPE_AGENT) < epoch)
                    __builtin_amdgcn_s_sleep(1);
            }
        }
        __syncthreads();
    }
}

// ---------------------------------------------------------------------------
// Kernel 3: out = h2_seq @ Wfc^T + bfc  (M=65536, N=512, K=512) bf16 WMMA
// block = 8 waves; wave w covers vocab cols [64w, 64w+64) of one 16-row tile.
// ---------------------------------------------------------------------------
extern "C" __global__ __launch_bounds__(256) void lstm_proj(
    const unsigned short* __restrict__ h2seq,
    const unsigned short* __restrict__ wfcb,
    const float* __restrict__ bfc, float* __restrict__ out)
{
    const int lane   = threadIdx.x & 31;
    const int wave   = threadIdx.x >> 5;     // n-group 0..7
    const int mrow16 = blockIdx.x * 16;      // 0..65520

    v8f acc[4];
#pragma unroll
    for (int nt = 0; nt < 4; ++nt) {
        float bi = bfc[wave * 64 + nt * 16 + (lane & 15)];
#pragma unroll
        for (int r = 0; r < 8; ++r) acc[nt][r] = bi;
    }
#pragma unroll 4
    for (int kb = 0; kb < 16; ++kb) {
        v16bf a = load_mat(h2seq, HID, mrow16, lane, kb);
#pragma unroll
        for (int nt = 0; nt < 4; ++nt) {
            v16bf b = load_mat(wfcb, HID, wave * 64 + nt * 16, lane, kb);
            acc[nt] = wmma_bf16(a, b, acc[nt]);
        }
    }
    const int mrow0 = mrow16 + ((lane >> 4) << 3);
#pragma unroll
    for (int nt = 0; nt < 4; ++nt) {
        int v = wave * 64 + nt * 16 + (lane & 15);
#pragma unroll
        for (int r = 0; r < 8; ++r)
            out[(size_t)(mrow0 + r) * VOCAB + v] = acc[nt][r];
    }
}

// ---------------------------------------------------------------------------
extern "C" void kernel_launch(void* const* d_in, const int* in_sizes, int n_in,
                              void* d_out, int out_size, void* d_ws, size_t ws_size,
                              hipStream_t stream) {
    const int*   x    = (const int*)d_in[0];
    const float* h0   = (const float*)d_in[1];
    const float* c0   = (const float*)d_in[2];
    const float* emb  = (const float*)d_in[3];
    const float* Wih1 = (const float*)d_in[4];
    const float* Whh1 = (const float*)d_in[5];
    const float* bih1 = (const float*)d_in[6];
    const float* bhh1 = (const float*)d_in[7];
    const float* Wih2 = (const float*)d_in[8];
    const float* Whh2 = (const float*)d_in[9];
    const float* bih2 = (const float*)d_in[10];
    const float* bhh2 = (const float*)d_in[11];
    const float* Wfc  = (const float*)d_in[12];
    const float* bfc  = (const float*)d_in[13];

    char* ws = (char*)d_ws;
    size_t off = 0;
    int* bar = (int*)(ws + off);                        off += 256;
    unsigned short* xe    = (unsigned short*)(ws + off); off += (size_t)SEQL * BATCH * EMBD * 2;
    unsigned short* h1s   = (unsigned short*)(ws + off); off += (size_t)2 * BATCH * HID * 2;
    unsigned short* h2s   = (unsigned short*)(ws + off); off += (size_t)2 * BATCH * HID * 2;
    unsigned short* h2seq = (unsigned short*)(ws + off); off += (size_t)SEQL * BATCH * HID * 2;
    unsigned short* wfcb  = (unsigned short*)(ws + off); off += (size_t)VOCAB * HID * 2;
    float* b1 = (float*)(ws + off);                      off += 4 * HID * 4;
    float* b2 = (float*)(ws + off);                      off += 4 * HID * 4;

    (void)hipFuncSetAttribute((const void*)lstm_recurrent,
                              hipFuncAttributeMaxDynamicSharedMemorySize, 229376);

    lstm_prep<<<8192, 256, 0, stream>>>(x, h0, emb, bih1, bhh1, bih2, bhh2, Wfc,
                                        xe, h1s, h2s, wfcb, b1, b2, bar);
    lstm_recurrent<<<NBLK, 128, 229376, stream>>>(c0, xe, Wih1, Whh1, Wih2, Whh2,
                                                  b1, b2, h1s, h2s, h2seq, bar);
    lstm_proj<<<(SEQL * BATCH) / 16, 256, 0, stream>>>(h2seq, wfcb, bfc, (float*)d_out);
}
